// HashTimeRadianceField_47141561041215
// MI455X (gfx1250) — matrix operand order repaired and checked by reading, not compiled, not run on hardware
//
#include <hip/hip_runtime.h>
#include <hip/hip_bf16.h>

typedef __attribute__((ext_vector_type(16))) _Float16 v16h;
typedef __attribute__((ext_vector_type(8)))  float    v8f;
typedef __attribute__((ext_vector_type(4)))  unsigned v4u;

#define LVLS   16
#define TSIZE  (1u << 19)          // hash table entries per level
#define TILE   256                 // points per block

// LDS layout (in halves)
#define W1_OFF   0                 // [64][32]
#define W2_OFF   2048              // [16][64]
#define WC1_OFF  3072              // [64][32]
#define WC2_OFF  5120              // [64][64]
#define WC3_OFF  9216              // [16][64] (rows 3..15 zero)
#define W_TOTAL  10240
#define CIST     40                // ci stride: [256][32] + pad
#define AST      72                // act stride: [256][64] + pad

struct A32 { v4u lo; v4u hi; };

// ---- WMMA fragment helpers (ISA 7.12.2 layouts, wave32) --------------------

// A (16x32 f16): lanes 0-15 hold row M=lane, K {kb+0..7, kb+16..23};
//                lanes 16-31 hold row M=lane-16, K {kb+8..15, kb+24..31}
__device__ __forceinline__ v16h load_a_frag(const _Float16* s, int stride, int m0, int kb) {
    int lane = threadIdx.x & 31;
    int row  = m0 + (lane & 15);
    int kh   = (lane >> 4) * 8;
    const _Float16* p = s + row * stride + kb + kh;
    v4u lo = *(const v4u*)(p);
    v4u hi = *(const v4u*)(p + 16);
    return __builtin_bit_cast(v16h, A32{lo, hi});
}

// B (32x16 f16) from row-major W[out][K]: lane n%16 = output column,
// lanes 0-15 hold K {kb+0..7, kb+16..23}, lanes 16-31 hold K {kb+8..15, kb+24..31}
__device__ __forceinline__ v16h load_b_frag(const _Float16* w, int K, int n0, int kb) {
    int lane = threadIdx.x & 31;
    int col  = n0 + (lane & 15);
    int ko   = (lane >> 4) * 8;
    const _Float16* p = w + col * K + kb + ko;
    v4u lo = *(const v4u*)(p);
    v4u hi = *(const v4u*)(p + 16);
    return __builtin_bit_cast(v16h, A32{lo, hi});
}

// D (16x16 f32): VGPR r -> M = r + 8*(lane>=16), N = lane%16
__device__ __forceinline__ void store_d_f16(_Float16* s, int stride, int m0, int n0,
                                            v8f d, bool relu) {
    int lane = threadIdx.x & 31;
    int n    = lane & 15;
    int mb   = (lane >> 4) * 8;
#pragma unroll
    for (int r = 0; r < 8; ++r) {
        float v = d[r];
        if (relu) v = v > 0.0f ? v : 0.0f;
        s[(m0 + mb + r) * stride + n0 + n] = (_Float16)v;
    }
}

__device__ __forceinline__ v8f wmma16(v16h a, v16h b, v8f c) {
    return __builtin_amdgcn_wmma_f32_16x16x32_f16(false, a, false, b, (short)0, c,
                                                  false, false);
}

// ---------------------------------------------------------------------------

extern "C" __global__ __launch_bounds__(256)
void ngp_field_kernel(const float* __restrict__ xyzs,   // [N,4]
                      const float* __restrict__ dirs,   // [N,3]
                      const float* __restrict__ grid,   // [16, 2^19, 2]
                      const float* __restrict__ W1,     // [64,32]
                      const float* __restrict__ W2,     // [16,64]
                      const float* __restrict__ Wc1,    // [64,32]
                      const float* __restrict__ Wc2,    // [64,64]
                      const float* __restrict__ Wc3,    // [3,64]
                      float* __restrict__ out_color,    // [N,3]
                      float* __restrict__ out_sigma)    // [N]
{
    extern __shared__ char smem_raw[];
    _Float16* s_w  = (_Float16*)smem_raw;     // W_TOTAL
    _Float16* s_ci = s_w  + W_TOTAL;          // [256][CIST]  sh | h
    _Float16* s_a  = s_ci + TILE * CIST;      // [256][AST]   hr, later C1
    _Float16* s_b  = s_a  + TILE * AST;       // [256][AST]   H1, later C2

    const int tid  = threadIdx.x;
    const int wave = tid >> 5;
    const int lane = tid & 31;
    const long pt  = (long)blockIdx.x * TILE + tid;

    // ---- Phase 0: weights f32 -> f16 in LDS -------------------------------
    for (int i = tid; i < 2048; i += 256) s_w[W1_OFF  + i] = (_Float16)W1[i];
    for (int i = tid; i < 1024; i += 256) s_w[W2_OFF  + i] = (_Float16)W2[i];
    for (int i = tid; i < 2048; i += 256) s_w[WC1_OFF + i] = (_Float16)Wc1[i];
    for (int i = tid; i < 4096; i += 256) s_w[WC2_OFF + i] = (_Float16)Wc2[i];
    for (int i = tid; i < 1024; i += 256)
        s_w[WC3_OFF + i] = (_Float16)(i < 192 ? Wc3[i] : 0.0f);

    // ---- Phase 1a: spherical harmonics -> s_ci cols 0..15 -----------------
    {
        float x = dirs[3 * pt + 0] * 2.0f - 1.0f;
        float y = dirs[3 * pt + 1] * 2.0f - 1.0f;
        float z = dirs[3 * pt + 2] * 2.0f - 1.0f;
        float xy = x * y, xz = x * z, yz = y * z;
        float x2 = x * x, y2 = y * y, z2 = z * z;
        float sh[16];
        sh[0]  = 0.28209479177387814f;
        sh[1]  = -0.48860251190291987f * y;
        sh[2]  =  0.48860251190291987f * z;
        sh[3]  = -0.48860251190291987f * x;
        sh[4]  =  1.0925484305920792f * xy;
        sh[5]  = -1.0925484305920792f * yz;
        sh[6]  =  0.94617469575756f * z2 - 0.31539156525252f;
        sh[7]  = -1.0925484305920792f * xz;
        sh[8]  =  0.5462742152960396f * (x2 - y2);
        sh[9]  =  0.5900435899266435f * y * (-3.0f * x2 + y2);
        sh[10] =  2.890611442640554f * xy * z;
        sh[11] =  0.4570457994644657f * y * (1.0f - 5.0f * z2);
        sh[12] =  0.3731763325901154f * z * (5.0f * z2 - 3.0f);
        sh[13] =  0.4570457994644657f * x * (1.0f - 5.0f * z2);
        sh[14] =  1.445305721320277f * z * (x2 - y2);
        sh[15] =  0.5900435899266435f * x * (-x2 + 3.0f * y2);
#pragma unroll
        for (int j = 0; j < 16; ++j)
            s_ci[tid * CIST + j] = (_Float16)sh[j];
    }

    // ---- Phase 1b: 4-D multires hash encoding -> s_a cols 0..31 -----------
    {
        const float4 pv = ((const float4*)xyzs)[pt];
        float x[4];
        x[0] = (pv.x + 1.0f) * 0.5f;
        x[1] = (pv.y + 1.0f) * 0.5f;
        x[2] = (pv.z + 1.0f) * 0.5f;
        x[3] = pv.w;

#pragma unroll 1
        for (int l = 0; l < LVLS; ++l) {
            float scale = exp2f((float)l * (8.0f / 15.0f)) * 8.0f - 1.0f;
            int   res   = (int)ceilf(scale) + 1;
            bool  dense = ((long long)res * res * res * res <= (long long)TSIZE);

            float fr[4]; int gi[4];
#pragma unroll
            for (int d = 0; d < 4; ++d) {
                float p = x[d] * scale + 0.5f;
                float g = floorf(p);
                fr[d] = p - g;
                gi[d] = (int)g;
            }

            float f0 = 0.0f, f1 = 0.0f;
            const float* gl = grid + ((size_t)l << 20);   // l*T*F
#pragma unroll
            for (int c = 0; c < 16; ++c) {
                float w = 1.0f;
                unsigned cx[4];
#pragma unroll
                for (int d = 0; d < 4; ++d) {
                    int b = (c >> d) & 1;
                    w *= b ? fr[d] : (1.0f - fr[d]);
                    cx[d] = (unsigned)(gi[d] + b);
                }
                unsigned idx;
                if (dense) {
                    unsigned lin = cx[0] + (unsigned)res *
                                   (cx[1] + (unsigned)res *
                                   (cx[2] + (unsigned)res * cx[3]));
                    idx = lin < (TSIZE - 1u) ? lin : (TSIZE - 1u);
                } else {
                    idx = (cx[0] * 1u) ^ (cx[1] * 2654435761u) ^
                          (cx[2] * 805459861u) ^ (cx[3] * 3674351687u);
                    idx &= (TSIZE - 1u);
                }
                const float* gp = gl + (size_t)idx * 2u;
                f0 += w * gp[0];
                f1 += w * gp[1];
            }
            s_a[tid * AST + 2 * l + 0] = (_Float16)f0;
            s_a[tid * AST + 2 * l + 1] = (_Float16)f1;
        }
    }
    __syncthreads();

    // ---- Layer 1: H1 = relu(hr @ W1^T)  [256,32]x[32,64] -> s_b -----------
#pragma unroll
    for (int t = 0; t < 2; ++t) {
        int m0 = (wave * 2 + t) * 16;
        v16h a = load_a_frag(s_a, AST, m0, 0);
#pragma unroll
        for (int nt = 0; nt < 4; ++nt) {
            v8f acc = {};
            acc = wmma16(a, load_b_frag(s_w + W1_OFF, 32, nt * 16, 0), acc);
            store_d_f16(s_b, AST, m0, nt * 16, acc, true);
        }
    }
    __syncthreads();

    // ---- Layer 2: h = H1 @ W2^T  [256,64]x[64,16]; sigma = exp(h[:,0]) ----
#pragma unroll
    for (int t = 0; t < 2; ++t) {
        int m0 = (wave * 2 + t) * 16;
        v8f acc = {};
#pragma unroll
        for (int kc = 0; kc < 2; ++kc)
            acc = wmma16(load_a_frag(s_b, AST, m0, kc * 32),
                         load_b_frag(s_w + W2_OFF, 64, 0, kc * 32), acc);
        store_d_f16(s_ci, CIST, m0, 16, acc, false);   // ci cols 16..31 = h
        if ((lane & 15) == 0) {                        // column n == 0 lanes
            int mb = (lane >> 4) * 8;
#pragma unroll
            for (int r = 0; r < 8; ++r)
                out_sigma[(long)blockIdx.x * TILE + m0 + mb + r] = expf(acc[r]);
        }
    }
    __syncthreads();

    // ---- Color layer 1: C1 = relu(ci @ Wc1^T) -> s_a -----------------------
#pragma unroll
    for (int t = 0; t < 2; ++t) {
        int m0 = (wave * 2 + t) * 16;
        v16h a = load_a_frag(s_ci, CIST, m0, 0);
#pragma unroll
        for (int nt = 0; nt < 4; ++nt) {
            v8f acc = {};
            acc = wmma16(a, load_b_frag(s_w + WC1_OFF, 32, nt * 16, 0), acc);
            store_d_f16(s_a, AST, m0, nt * 16, acc, true);
        }
    }
    __syncthreads();

    // ---- Color layer 2: C2 = relu(C1 @ Wc2^T) -> s_b -----------------------
#pragma unroll
    for (int t = 0; t < 2; ++t) {
        int m0 = (wave * 2 + t) * 16;
#pragma unroll
        for (int nt = 0; nt < 4; ++nt) {
            v8f acc = {};
#pragma unroll
            for (int kc = 0; kc < 2; ++kc)
                acc = wmma16(load_a_frag(s_a, AST, m0, kc * 32),
                             load_b_frag(s_w + WC2_OFF, 64, nt * 16, kc * 32), acc);
            store_d_f16(s_b, AST, m0, nt * 16, acc, true);
        }
    }
    __syncthreads();

    // ---- Color layer 3: color = sigmoid(C2 @ Wc3pad^T)[:, :3] --------------
#pragma unroll
    for (int t = 0; t < 2; ++t) {
        int m0 = (wave * 2 + t) * 16;
        v8f acc = {};
#pragma unroll
        for (int kc = 0; kc < 2; ++kc)
            acc = wmma16(load_a_frag(s_b, AST, m0, kc * 32),
                         load_b_frag(s_w + WC3_OFF, 64, 0, kc * 32), acc);
        int n  = lane & 15;
        int mb = (lane >> 4) * 8;
        if (n < 3) {
#pragma unroll
            for (int r = 0; r < 8; ++r) {
                long row = (long)blockIdx.x * TILE + m0 + mb + r;
                float v  = acc[r];
                out_color[row * 3 + n] = 1.0f / (1.0f + expf(-v));
            }
        }
    }
}

extern "C" void kernel_launch(void* const* d_in, const int* in_sizes, int n_in,
                              void* d_out, int out_size, void* d_ws, size_t ws_size,
                              hipStream_t stream) {
    (void)n_in; (void)d_ws; (void)ws_size; (void)out_size;
    const float* xyzs = (const float*)d_in[0];
    const float* dirs = (const float*)d_in[1];
    const float* grid = (const float*)d_in[2];
    const float* W1   = (const float*)d_in[3];
    const float* W2   = (const float*)d_in[4];
    const float* Wc1  = (const float*)d_in[5];
    const float* Wc2  = (const float*)d_in[6];
    const float* Wc3  = (const float*)d_in[7];
    float* out = (float*)d_out;

    const int N = in_sizes[0] / 4;                 // 262144
    const int blocks = N / TILE;                   // 1024
    const size_t smem = (size_t)(W_TOTAL + TILE * CIST + 2 * TILE * AST) * sizeof(_Float16);

    ngp_field_kernel<<<blocks, TILE, smem, stream>>>(
        xyzs, dirs, grid, W1, W2, Wc1, Wc2, Wc3,
        out,                     // color [N,3]
        out + (size_t)N * 3);    // sigma [N]
}
